// CausalSelfAttention_58703613001764
// MI455X (gfx1250) — matrix operand order repaired and checked by reading, not compiled, not run on hardware
//
#include <hip/hip_runtime.h>
#include <hip/hip_bf16.h>

// ---------------------------------------------------------------------------
// CausalSelfAttention on MI455X (gfx1250, wave32, WMMA f16 -> f32 accum)
//   B=4, T=2048, C=1024, H=16, D=64
// Pipeline:
//   1) x (f32) -> xh (f16)
//   2) W_qkv, W_proj (f32, KxN) -> N-major f16 (N x K)
//   3) QKV GEMM (WMMA, async-LDS double-buffered), scatter Q/K/Vt
//   4) Flash attention (WMMA QK^T + online softmax + WMMA PV)
//   5) proj GEMM (WMMA) + bias -> f32 out
// ---------------------------------------------------------------------------

typedef __attribute__((ext_vector_type(16))) _Float16 v16h;
typedef __attribute__((ext_vector_type(8)))  _Float16 v8h;
typedef __attribute__((ext_vector_type(4)))  _Float16 v4h;
typedef __attribute__((ext_vector_type(8)))  float    v8f;

#define BATCH   4
#define SEQ     2048
#define EMB     1024
#define NHEAD   16
#define HDIM    64
#define QKVN    3072
#define MTOT    (BATCH*SEQ)      // 8192

__device__ inline v16h cat16(v8h lo, v8h hi) {
  return __builtin_shufflevector(lo, hi, 0,1,2,3,4,5,6,7,8,9,10,11,12,13,14,15);
}

__device__ inline v8f wmma_f16(v16h a, v16h b, v8f c) {
  // D = A(16x32 f16) * B(32x16 f16) + C(16x16 f32)
  return __builtin_amdgcn_wmma_f32_16x16x32_f16(
      /*neg_a=*/false, a, /*neg_b=*/false, b,
      /*c_mod=*/(short)0, c, /*reuse_a=*/false, /*reuse_b=*/false);
}

// Async 16-byte copy: global memory -> LDS, tracked by ASYNCcnt.
// LDS address = low 32 bits of the flat address (aperture truncation).
__device__ inline void async_cp128(const _Float16* g, _Float16* l) {
  uint32_t lds_addr = (uint32_t)(uintptr_t)l;
  asm volatile("global_load_async_to_lds_b128 %0, %1, off"
               :: "v"(lds_addr), "v"(g) : "memory");
}

// ---------------------------------------------------------------------------
// elementwise converts
// ---------------------------------------------------------------------------
__global__ void cvt_f32_to_f16_x4(const float* __restrict__ in,
                                  _Float16* __restrict__ out, int n4) {
  int i = blockIdx.x * blockDim.x + threadIdx.x;
  if (i < n4) {
    float4 v = ((const float4*)in)[i];
    v4h o;
    o[0] = (_Float16)v.x; o[1] = (_Float16)v.y;
    o[2] = (_Float16)v.z; o[3] = (_Float16)v.w;
    ((v4h*)out)[i] = o;
  }
}

// in: (Kdim x Ndim) row-major f32  ->  out: (Ndim x Kdim) row-major f16
__global__ void transpose_cvt_f16(const float* __restrict__ in,
                                  _Float16* __restrict__ out,
                                  int Kdim, int Ndim) {
  int idx = blockIdx.x * blockDim.x + threadIdx.x;
  if (idx < Kdim * Ndim) {
    int n = idx / Kdim;
    int k = idx - n * Kdim;
    out[idx] = (_Float16)in[k * Ndim + n];
  }
}

// ---------------------------------------------------------------------------
// Shared 128x128 WMMA tile engine, async-LDS double buffered.
// A: row-major (M x K) f16, lda.  B: N-major (N x K) f16, ldb.
// Block: 256 threads = 8 waves, wave grid 4(M) x 2(N); wave owns 32x64.
// LDS rows padded to 40 f16 (80B): 16B alignment kept for b128 async copies.
// ---------------------------------------------------------------------------
#define LDSROW 40
#define TILEHALF (128 * LDSROW)    // one buffer of one matrix, in f16 elems

__device__ inline void gemm_mainloop(const _Float16* __restrict__ Ag,
                                     const _Float16* __restrict__ Bg,
                                     int lda, int ldb, int Ktot,
                                     int mBase, int nBase,
                                     _Float16* sA, _Float16* sB,  // [2][TILEHALF]
                                     v8f acc[2][4]) {
  const int tid   = threadIdx.x;
  const int lane  = tid & 31;
  const int wv    = tid >> 5;
  const int waveM = wv >> 1;       // 0..3
  const int waveN = wv & 1;        // 0..1
  const int hi    = lane >> 4;     // lane half
  const int l16   = lane & 15;

  // this thread's two 8-f16 chunk slots (row, cc) for the 128x32 tiles
  const int row0 = tid >> 2,        cc0 = tid & 3;
  const int row1 = (tid + 256) >> 2, cc1 = tid & 3;

  // prologue: issue async copies for slice 0 into buffer 0
  {
    async_cp128(Ag + (size_t)(mBase + row0) * lda + cc0 * 8,
                sA + row0 * LDSROW + cc0 * 8);
    async_cp128(Bg + (size_t)(nBase + row0) * ldb + cc0 * 8,
                sB + row0 * LDSROW + cc0 * 8);
    async_cp128(Ag + (size_t)(mBase + row1) * lda + cc1 * 8,
                sA + row1 * LDSROW + cc1 * 8);
    async_cp128(Bg + (size_t)(nBase + row1) * ldb + cc1 * 8,
                sB + row1 * LDSROW + cc1 * 8);
  }

  const int nsteps = Ktot >> 5;    // K / 32
  for (int i = 0; i < nsteps; i++) {
    const int cur = i & 1;
    if (i + 1 < nsteps) {
      // issue async copies for the next slice into the other buffer
      const int nxt = (i + 1) & 1;
      const int k0  = (i + 1) * 32;
      _Float16* dA = sA + nxt * TILEHALF;
      _Float16* dB = sB + nxt * TILEHALF;
      async_cp128(Ag + (size_t)(mBase + row0) * lda + k0 + cc0 * 8,
                  dA + row0 * LDSROW + cc0 * 8);
      async_cp128(Bg + (size_t)(nBase + row0) * ldb + k0 + cc0 * 8,
                  dB + row0 * LDSROW + cc0 * 8);
      async_cp128(Ag + (size_t)(mBase + row1) * lda + k0 + cc1 * 8,
                  dA + row1 * LDSROW + cc1 * 8);
      async_cp128(Bg + (size_t)(nBase + row1) * ldb + k0 + cc1 * 8,
                  dB + row1 * LDSROW + cc1 * 8);
      // current slice = the 4 older outstanding async ops (in-order retire)
      asm volatile("s_wait_asynccnt 0x4" ::: "memory");
    } else {
      asm volatile("s_wait_asynccnt 0x0" ::: "memory");
    }
    __syncthreads();               // all waves' current slice landed in LDS

    const _Float16* cA = sA + cur * TILEHALF;
    const _Float16* cB = sB + cur * TILEHALF;
    v16h aF[2], bF[4];
#pragma unroll
    for (int mt = 0; mt < 2; mt++) {
      // A frag: lane holds row M=l16; f16 elems i<8 -> k=i+8*hi, i>=8 -> +16
      const _Float16* p = &cA[(waveM * 32 + mt * 16 + l16) * LDSROW + 8 * hi];
      aF[mt] = cat16(*(const v8h*)p, *(const v8h*)(p + 16));
    }
#pragma unroll
    for (int nt = 0; nt < 4; nt++) {
      // B frag: lane holds col N=l16; f16 elems i<8 -> k=i+8*hi, i>=8 -> +16
      const _Float16* p = &cB[(waveN * 64 + nt * 16 + l16) * LDSROW + 8 * hi];
      bF[nt] = cat16(*(const v8h*)p, *(const v8h*)(p + 16));
    }
#pragma unroll
    for (int mt = 0; mt < 2; mt++)
#pragma unroll
      for (int nt = 0; nt < 4; nt++)
        acc[mt][nt] = wmma_f16(aF[mt], bF[nt], acc[mt][nt]);

    __syncthreads();               // protect buffer overwritten next iteration
  }
}

// ---------------------------------------------------------------------------
// QKV GEMM: xh (8192 x 1024) @ WqkvT (3072 x 1024, N-major) + bias
// Epilogue: scatter into Q (B,H,T,D) *1/sqrt(D), K (B,H,T,D), Vt (B,H,D,T)
// ---------------------------------------------------------------------------
__global__ __launch_bounds__(256) void qkv_gemm_kernel(
    const _Float16* __restrict__ xh, const _Float16* __restrict__ wT,
    const float* __restrict__ bias,
    _Float16* __restrict__ Qb, _Float16* __restrict__ Kb,
    _Float16* __restrict__ Vt) {
  __shared__ __align__(16) _Float16 sA[2 * TILEHALF];
  __shared__ __align__(16) _Float16 sB[2 * TILEHALF];
  const int mBase = blockIdx.x * 128;
  const int nBase = blockIdx.y * 128;
  v8f acc[2][4] = {};
  gemm_mainloop(xh, wT, EMB, EMB, EMB, mBase, nBase, sA, sB, acc);

  const int tid = threadIdx.x, lane = tid & 31, wv = tid >> 5;
  const int waveM = wv >> 1, waveN = wv & 1;
  const int hi = lane >> 4, l16 = lane & 15;
#pragma unroll
  for (int mt = 0; mt < 2; mt++)
#pragma unroll
    for (int nt = 0; nt < 4; nt++)
#pragma unroll
      for (int r = 0; r < 8; r++) {
        int m = mBase + waveM * 32 + mt * 16 + r + 8 * hi;   // token row
        int n = nBase + waveN * 64 + nt * 16 + l16;          // channel 0..3071
        float v = acc[mt][nt][r] + bias[n];
        int b = m >> 11, t = m & (SEQ - 1);
        if (n < EMB) {
          int h = n >> 6, d = n & 63;
          Qb[(((size_t)(b * NHEAD + h)) * SEQ + t) * HDIM + d] =
              (_Float16)(v * 0.125f);                        // 1/sqrt(64)
        } else if (n < 2 * EMB) {
          int n2 = n - EMB; int h = n2 >> 6, d = n2 & 63;
          Kb[(((size_t)(b * NHEAD + h)) * SEQ + t) * HDIM + d] = (_Float16)v;
        } else {
          int n2 = n - 2 * EMB; int h = n2 >> 6, d = n2 & 63;
          Vt[(((size_t)(b * NHEAD + h)) * HDIM + d) * SEQ + t] = (_Float16)v;
        }
      }
}

// ---------------------------------------------------------------------------
// Flash attention. Block = 128 thr = 4 waves, each wave: 16 query rows.
// Key blocks of 32. S = two 16x16 C-tiles; P through per-wave LDS patch.
// ---------------------------------------------------------------------------
__global__ __launch_bounds__(128) void attn_kernel(
    const _Float16* __restrict__ Qb, const _Float16* __restrict__ Kb,
    const _Float16* __restrict__ Vt, _Float16* __restrict__ A2) {
  __shared__ __align__(16) _Float16 sP[4][16 * LDSROW];
  const int tid = threadIdx.x, wv = tid >> 5, lane = tid & 31;
  const int hi = lane >> 4, l16 = lane & 15;

  const int blk = blockIdx.x;         // 0 .. B*H*T/64 - 1
  const int qblock = blk & 31;        // T/64 = 32
  const int bh = blk >> 5;            // 0..63
  const int qb = qblock * 64 + wv * 16;

  const _Float16* Q  = Qb + (size_t)bh * SEQ * HDIM;
  const _Float16* Kp = Kb + (size_t)bh * SEQ * HDIM;
  const _Float16* Vp = Vt + (size_t)bh * HDIM * SEQ;

  // Q fragments for the two 32-wide halves of D
  v16h qf[2];
  {
    int qrow = qb + l16;
#pragma unroll
    for (int h2 = 0; h2 < 2; h2++) {
      const _Float16* p = Q + (size_t)qrow * HDIM + h2 * 32 + 8 * hi;
      qf[h2] = cat16(*(const v8h*)p, *(const v8h*)(p + 16));
    }
  }

  v8f oacc[4] = {};
  v8f mrow, lrow;
#pragma unroll
  for (int r = 0; r < 8; r++) { mrow[r] = -1e30f; lrow[r] = 0.0f; }

  const int kmaxRow = qb + 15;        // highest query row this wave owns
  for (int kb = 0; kb <= kmaxRow; kb += 32) {
    // prefetch next key/value block (global_prefetch_b8)
    if (kb + 32 <= kmaxRow) {
      __builtin_prefetch(Kp + (size_t)(kb + 32 + l16) * HDIM, 0, 3);
      __builtin_prefetch(Vp + (size_t)l16 * SEQ + kb + 32, 0, 3);
    }
    // ---- S = Q @ K^T (scaled Q already) : two 16-col tiles ----
    v8f s[2] = {};
#pragma unroll
    for (int tn = 0; tn < 2; tn++) {
      int srow = kb + tn * 16 + l16;
#pragma unroll
      for (int h2 = 0; h2 < 2; h2++) {
        const _Float16* p = Kp + (size_t)srow * HDIM + h2 * 32 + 8 * hi;
        v16h kf = cat16(*(const v8h*)p, *(const v8h*)(p + 16));
        s[tn] = wmma_f16(qf[h2], kf, s[tn]);
      }
    }
    // ---- causal mask (only near the diagonal) ----
    if (kb + 31 > qb) {
#pragma unroll
      for (int tn = 0; tn < 2; tn++)
#pragma unroll
        for (int r = 0; r < 8; r++) {
          int m = qb + r + 8 * hi;
          int n = kb + tn * 16 + l16;
          if (n > m) s[tn][r] = -1e30f;
        }
    }
    // ---- online softmax: row max / rescale / exp / row sum ----
    v8f tmax;
#pragma unroll
    for (int r = 0; r < 8; r++) tmax[r] = fmaxf(s[0][r], s[1][r]);
#pragma unroll
    for (int off = 1; off < 16; off <<= 1)
#pragma unroll
      for (int r = 0; r < 8; r++)
        tmax[r] = fmaxf(tmax[r], __shfl_xor(tmax[r], off, 32));
    v8f newm, alpha;
#pragma unroll
    for (int r = 0; r < 8; r++) {
      newm[r]  = fmaxf(mrow[r], tmax[r]);
      alpha[r] = __expf(mrow[r] - newm[r]);
      mrow[r]  = newm[r];
    }
#pragma unroll
    for (int tn = 0; tn < 2; tn++)
#pragma unroll
      for (int r = 0; r < 8; r++)
        s[tn][r] = __expf(s[tn][r] - newm[r]);
    v8f rs;
#pragma unroll
    for (int r = 0; r < 8; r++) rs[r] = s[0][r] + s[1][r];
#pragma unroll
    for (int off = 1; off < 16; off <<= 1)
#pragma unroll
      for (int r = 0; r < 8; r++) rs[r] += __shfl_xor(rs[r], off, 32);
#pragma unroll
    for (int r = 0; r < 8; r++) lrow[r] = lrow[r] * alpha[r] + rs[r];
#pragma unroll
    for (int dt = 0; dt < 4; dt++)
#pragma unroll
      for (int r = 0; r < 8; r++) oacc[dt][r] *= alpha[r];

    // ---- P (C-layout f32) -> A-layout f16 via per-wave LDS patch ----
    _Float16* pw = &sP[wv][0];
#pragma unroll
    for (int tn = 0; tn < 2; tn++)
#pragma unroll
      for (int r = 0; r < 8; r++)
        pw[(r + 8 * hi) * LDSROW + tn * 16 + l16] = (_Float16)s[tn][r];
    // per-wave LDS ordering: DS ops from one wave retire in order
    asm volatile("s_wait_dscnt 0x0" ::: "memory");
    v16h pf;
    {
      const _Float16* p = &pw[l16 * LDSROW + 8 * hi];
      pf = cat16(*(const v8h*)p, *(const v8h*)(p + 16));
    }
    // ---- O += P @ V  (V pre-transposed: (d, t)) ----
#pragma unroll
    for (int dt = 0; dt < 4; dt++) {
      const _Float16* p = Vp + (size_t)(dt * 16 + l16) * SEQ + kb + 8 * hi;
      v16h vf = cat16(*(const v8h*)p, *(const v8h*)(p + 16));
      oacc[dt] = wmma_f16(pf, vf, oacc[dt]);
    }
  }

  // ---- finalize + store to (B,T,C) f16 ----
  const int b = bh >> 4, h = bh & 15;
#pragma unroll
  for (int r = 0; r < 8; r++) {
    float inv = 1.0f / lrow[r];
    int t = qb + r + 8 * hi;
#pragma unroll
    for (int dt = 0; dt < 4; dt++) {
      int ch = h * HDIM + dt * 16 + l16;
      A2[((size_t)(b * SEQ + t)) * EMB + ch] = (_Float16)(oacc[dt][r] * inv);
    }
  }
}

// ---------------------------------------------------------------------------
// Output projection: A2 (8192 x 1024) @ WprojT (1024 x 1024, N-major) + bias
// ---------------------------------------------------------------------------
__global__ __launch_bounds__(256) void proj_gemm_kernel(
    const _Float16* __restrict__ A2, const _Float16* __restrict__ wT,
    const float* __restrict__ bias, float* __restrict__ out) {
  __shared__ __align__(16) _Float16 sA[2 * TILEHALF];
  __shared__ __align__(16) _Float16 sB[2 * TILEHALF];
  const int mBase = blockIdx.x * 128;
  const int nBase = blockIdx.y * 128;
  v8f acc[2][4] = {};
  gemm_mainloop(A2, wT, EMB, EMB, EMB, mBase, nBase, sA, sB, acc);

  const int tid = threadIdx.x, lane = tid & 31, wv = tid >> 5;
  const int waveM = wv >> 1, waveN = wv & 1;
  const int hi = lane >> 4, l16 = lane & 15;
#pragma unroll
  for (int mt = 0; mt < 2; mt++)
#pragma unroll
    for (int nt = 0; nt < 4; nt++)
#pragma unroll
      for (int r = 0; r < 8; r++) {
        int m = mBase + waveM * 32 + mt * 16 + r + 8 * hi;
        int n = nBase + waveN * 64 + nt * 16 + l16;
        out[(size_t)m * EMB + n] = acc[mt][nt][r] + bias[n];
      }
}

// ---------------------------------------------------------------------------
extern "C" void kernel_launch(void* const* d_in, const int* in_sizes, int n_in,
                              void* d_out, int out_size, void* d_ws,
                              size_t ws_size, hipStream_t stream) {
  (void)in_sizes; (void)n_in; (void)out_size; (void)ws_size;
  const float* x      = (const float*)d_in[0];
  const float* W_qkv  = (const float*)d_in[1];
  const float* b_qkv  = (const float*)d_in[2];
  const float* W_proj = (const float*)d_in[3];
  const float* b_proj = (const float*)d_in[4];
  float* out = (float*)d_out;

  char* ws = (char*)d_ws;
  size_t off = 0;
  auto take = [&](size_t bytes) {
    char* p = ws + off;
    off += (bytes + 255) & ~(size_t)255;
    return p;
  };
  _Float16* xh     = (_Float16*)take((size_t)MTOT * EMB * 2);     // 16 MB
  _Float16* wqkvT  = (_Float16*)take((size_t)QKVN * EMB * 2);     //  6 MB
  _Float16* wprojT = (_Float16*)take((size_t)EMB * EMB * 2);      //  2 MB
  _Float16* Qb     = (_Float16*)take((size_t)MTOT * EMB * 2);     // 16 MB
  _Float16* Kb     = (_Float16*)take((size_t)MTOT * EMB * 2);     // 16 MB
  _Float16* Vt     = (_Float16*)take((size_t)MTOT * EMB * 2);     // 16 MB
  _Float16* A2     = (_Float16*)take((size_t)MTOT * EMB * 2);     // 16 MB

  // 1) converts
  {
    int n4 = (MTOT * EMB) / 4;                          // 2,097,152
    cvt_f32_to_f16_x4<<<n4 / 256, 256, 0, stream>>>(x, xh, n4);
  }
  {
    int n = QKVN * EMB;
    transpose_cvt_f16<<<n / 256, 256, 0, stream>>>(W_qkv, wqkvT, EMB, QKVN);
  }
  {
    int n = EMB * EMB;
    transpose_cvt_f16<<<n / 256, 256, 0, stream>>>(W_proj, wprojT, EMB, EMB);
  }
  // 2) QKV GEMM + scatter
  qkv_gemm_kernel<<<dim3(MTOT / 128, QKVN / 128), 256, 0, stream>>>(
      xh, wqkvT, b_qkv, Qb, Kb, Vt);
  // 3) attention
  attn_kernel<<<BATCH * NHEAD * (SEQ / 64), 128, 0, stream>>>(Qb, Kb, Vt, A2);
  // 4) projection
  proj_gemm_kernel<<<dim3(MTOT / 128, EMB / 128), 256, 0, stream>>>(
      A2, wprojT, b_proj, out);
}
